// SARViGBranch_8383776161887
// MI455X (gfx1250) — matrix operand order repaired and checked by reading, compiled, not run on hardware
//
#include <hip/hip_runtime.h>
#include <hip/hip_bf16.h>

// ---------------------------------------------------------------------------
// SAR-ViG branch for MI455X (gfx1250), wave32 + v_wmma_f32_16x16x32_f16.
// GEMMs: WMMA f16 operands / f32 accumulation, 32x64 register-blocked
// macro-tiles per wave (8 WMMA per 12 b128 fragment loads). Elementwise
// stages (LN, top-k, max-relative aggregation, masked pool) are f32.
// node_mask (jnp.bool_) is treated as 1-byte per element.
// ---------------------------------------------------------------------------

typedef _Float16 h16_t;
typedef __attribute__((ext_vector_type(16))) _Float16 v16h;
typedef __attribute__((ext_vector_type(8)))  _Float16 v8h;
typedef __attribute__((ext_vector_type(8)))  float    v8f;

#define BB   256
#define NN   256
#define P2D  64
#define DD   128
#define LL   2
#define KNN  9
#define FFND 512
#define MROWS (BB * NN)   // 65536 token rows

// ---- WMMA fragment loaders (layouts per CDNA5 ISA §7.12.2) -----

// A-matrix 16x32 f16: lane L -> row M=L%16.
// halves 0..7  : K = k0 + (L/16)*8 + 0..7
// halves 8..15 : K = k0 + 16 + (L/16)*8 + 0..7
__device__ __forceinline__ v16h load_a_frag(const h16_t* Arow, int k0, int lane) {
  const int lh = (lane >> 4) * 8;
  v8h lo = *reinterpret_cast<const v8h*>(Arow + k0 + lh);
  v8h hi = *reinterpret_cast<const v8h*>(Arow + k0 + 16 + lh);
  v16h r;
#pragma unroll
  for (int i = 0; i < 8; ++i) { r[i] = lo[i]; r[i + 8] = hi[i]; }
  return r;
}

// B-matrix 32x16 f16 fed from a transposed [N,K] row: lane L -> col N=L%16.
// halves 0..7  : K = k0 + (L/16)*8 + 0..7
// halves 8..15 : K = k0 + 16 + (L/16)*8 + 0..7
__device__ __forceinline__ v16h load_b_frag(const h16_t* BTrow, int k0, int lane) {
  const int off = k0 + (lane >> 4) * 8;
  v8h lo = *reinterpret_cast<const v8h*>(BTrow + off);
  v8h hi = *reinterpret_cast<const v8h*>(BTrow + off + 16);
  v16h r;
#pragma unroll
  for (int i = 0; i < 8; ++i) { r[i] = lo[i]; r[i + 8] = hi[i]; }
  return r;
}

__device__ __forceinline__ v8f wmma_f16(v16h a, v16h b, v8f c) {
  return __builtin_amdgcn_wmma_f32_16x16x32_f16(false, a, false, b,
                                                (short)0, c, false, false);
}

// ---- Generic WMMA GEMM:  out = act( A[M,K] @ BT[N,K]^T + bias (+res) ) ----
// Register blocking: each wave computes a 32x64 tile (2 m-tiles x 4 n-tiles),
// reusing each A fragment 4x and each B fragment 2x. 4 waves per block.
// Requires M % 128 == 0, N % 64 == 0, K % 32 == 0 (all shapes here comply).
// act: 0 = none, 1 = leaky-relu(0.01). mask != null: zero row if !mask[m].
__global__ __launch_bounds__(128)
void k_gemm(const h16_t* __restrict__ A, const h16_t* __restrict__ BT,
            const float* __restrict__ bias, const float* __restrict__ res,
            const unsigned char* __restrict__ mask,
            float* __restrict__ out32, h16_t* __restrict__ out16,
            int M, int N, int Kd, int act)
{
  const int lane = threadIdx.x & 31;
  const int wave = threadIdx.x >> 5;
  const int m0 = (blockIdx.y * 4 + wave) * 32;   // 2 m-tiles per wave
  const int n0 = blockIdx.x * 64;                // 4 n-tiles per wave
  if (m0 >= M) return;

  const h16_t* Arow0 = A + (size_t)(m0 + (lane & 15)) * Kd;
  const h16_t* Arow1 = Arow0 + (size_t)16 * Kd;
  const h16_t* Brow[4];
#pragma unroll
  for (int j = 0; j < 4; ++j)
    Brow[j] = BT + (size_t)(n0 + 16 * j + (lane & 15)) * Kd;

  v8f c[2][4] = {};
  for (int k0 = 0; k0 < Kd; k0 += 32) {
    if (k0 + 64 <= Kd) {               // prefetch next-next A chunk
      __builtin_prefetch(Arow0 + k0 + 64, 0, 1);
      __builtin_prefetch(Arow1 + k0 + 64, 0, 1);
    }
    v16h a0 = load_a_frag(Arow0, k0, lane);
    v16h a1 = load_a_frag(Arow1, k0, lane);
#pragma unroll
    for (int j = 0; j < 4; ++j) {
      v16h b = load_b_frag(Brow[j], k0, lane);
      c[0][j] = wmma_f16(a0, b, c[0][j]);
      c[1][j] = wmma_f16(a1, b, c[1][j]);
    }
  }

#pragma unroll
  for (int j = 0; j < 4; ++j) {
    const int nn = n0 + 16 * j + (lane & 15);
    const float bv = bias ? bias[nn] : 0.0f;
#pragma unroll
    for (int i = 0; i < 2; ++i) {
#pragma unroll
      for (int v = 0; v < 8; ++v) {
        const int m = m0 + 16 * i + v + (lane >> 4) * 8;
        const size_t o = (size_t)m * N + nn;
        float val = c[i][j][v] + bv;
        if (res)      val += res[o];
        if (act == 1) val = (val >= 0.0f) ? val : 0.01f * val;
        if (mask)     val = mask[m] ? val : 0.0f;
        if (out32) out32[o] = val;
        if (out16) out16[o] = (h16_t)val;
      }
    }
  }
}

// ---- Batched Gram -> masked squared distances: d2[b,m,n] ------------------
// G[m,n] = sum_k h[b,m,k]*h[b,n,k]; d2 = sq[m]+sq[n]-2G; masked cols -> 1e10.
__global__ __launch_bounds__(128)
void k_gram_d2(const h16_t* __restrict__ h16, const float* __restrict__ sq,
               const unsigned char* __restrict__ mask, float* __restrict__ d2)
{
  const int lane = threadIdx.x & 31;
  const int wave = threadIdx.x >> 5;
  const int b  = blockIdx.z;
  const int m0 = (blockIdx.y * 4 + wave) * 16;
  const int n0 = blockIdx.x * 16;

  const h16_t* base = h16 + (size_t)b * NN * DD;
  const h16_t* Arow = base + (size_t)(m0 + (lane & 15)) * DD;
  const h16_t* Brow = base + (size_t)(n0 + (lane & 15)) * DD;

  v8f c = {};
  for (int k0 = 0; k0 < DD; k0 += 32) {
    v16h a = load_a_frag(Arow, k0, lane);
    v16h bb = load_b_frag(Brow, k0, lane);
    c = wmma_f16(a, bb, c);
  }

  const int n = n0 + (lane & 15);
  const float sqn = sq[b * NN + n];
  const bool mv = mask[b * NN + n] != 0;
#pragma unroll
  for (int v = 0; v < 8; ++v) {
    const int m = m0 + v + (lane >> 4) * 8;
    float dv = sq[b * NN + m] + sqn - 2.0f * c[v];
    if (!mv) dv = 1e10f;
    d2[((size_t)b * NN + m) * NN + n] = dv;
  }
}

// ---- Row LayerNorm (+optional lrelu, +optional row mask), wave per row ----
__global__ __launch_bounds__(128)
void k_ln(const float* __restrict__ X, const float* __restrict__ g,
          const float* __restrict__ bta, const unsigned char* __restrict__ mask,
          float* __restrict__ out32, h16_t* __restrict__ out16,
          int M, int Dn, int lrelu)
{
  const int lane = threadIdx.x & 31;
  const int row  = blockIdx.x * 4 + (threadIdx.x >> 5);
  if (row >= M) return;
  const float* x = X + (size_t)row * Dn;

  float s = 0.0f, s2 = 0.0f;
  for (int i = lane; i < Dn; i += 32) { float v = x[i]; s += v; s2 += v * v; }
  for (int off = 16; off; off >>= 1) {
    s  += __shfl_xor(s,  off, 32);
    s2 += __shfl_xor(s2, off, 32);
  }
  const float mean = s / Dn;
  const float inv  = rsqrtf(s2 / Dn - mean * mean + 1e-5f);
  const float mk = mask ? (mask[row] ? 1.0f : 0.0f) : 1.0f;

  for (int i = lane; i < Dn; i += 32) {
    float v = (x[i] - mean) * inv * g[i] + bta[i];
    if (lrelu) v = (v >= 0.0f) ? v : 0.01f * v;
    v *= mk;
    const size_t o = (size_t)row * Dn + i;
    if (out32) out32[o] = v;
    if (out16) out16[o] = (h16_t)v;
  }
}

// ---- per-row squared norm of h ----
__global__ void k_sqnorm(const float* __restrict__ h32, float* __restrict__ sq, int M) {
  const int r = blockIdx.x * blockDim.x + threadIdx.x;
  if (r >= M) return;
  const float* x = h32 + (size_t)r * DD;
  float s = 0.0f;
#pragma unroll 8
  for (int i = 0; i < DD; ++i) { float v = x[i]; s += v * v; }
  sq[r] = s;
}

// ---- top-K smallest per row (K=9), stable by index on ties ----
__global__ void k_topk(const float* __restrict__ d2, int* __restrict__ idx, int M) {
  const int r = blockIdx.x * blockDim.x + threadIdx.x;
  if (r >= M) return;
  const float* row = d2 + (size_t)r * NN;
  float bd[KNN]; int bi[KNN];
#pragma unroll
  for (int i = 0; i < KNN; ++i) { bd[i] = 3.4e38f; bi[i] = 0; }
  for (int n = 0; n < NN; ++n) {
    const float d = row[n];
    if (d < bd[KNN - 1]) {
      int j = KNN - 1;
      while (j > 0 && d < bd[j - 1]) { bd[j] = bd[j - 1]; bi[j] = bi[j - 1]; --j; }
      bd[j] = d; bi[j] = n;
    }
  }
#pragma unroll
  for (int i = 0; i < KNN; ++i) idx[(size_t)r * KNN + i] = bi[i];
}

// ---- max-relative aggregation; emits cat16 = [h, mrel] in f16 ----
__global__ __launch_bounds__(128)
void k_mrel(const float* __restrict__ h32, const int* __restrict__ idx,
            const unsigned char* __restrict__ mask, h16_t* __restrict__ cat16)
{
  const int row = blockIdx.x;      // token row 0..65535
  const int d   = threadIdx.x;     // 0..127
  const int b   = row / NN;
  const float hv = h32[(size_t)row * DD + d];
  float best = -1e30f; bool any = false;
#pragma unroll
  for (int k = 0; k < KNN; ++k) {
    const int j = idx[(size_t)row * KNN + k];
    if (mask[b * NN + j]) {
      any = true;
      const float val = h32[((size_t)b * NN + j) * DD + d] - hv;
      best = (val > best) ? val : best;
    }
  }
  h16_t* cr = cat16 + (size_t)row * (2 * DD);
  cr[d]      = (h16_t)hv;
  cr[DD + d] = (h16_t)(any ? best : 0.0f);
}

// ---- masked mean pool over nodes -> g16[b,d] ----
__global__ __launch_bounds__(128)
void k_pool(const float* __restrict__ x32, const unsigned char* __restrict__ mask,
            h16_t* __restrict__ g16)
{
  const int b = blockIdx.x, d = threadIdx.x;
  float s = 0.0f, cnt = 0.0f;
  for (int n = 0; n < NN; ++n) {
    const float mk = mask[b * NN + n] ? 1.0f : 0.0f;
    s   += x32[((size_t)b * NN + n) * DD + d] * mk;
    cnt += mk;
  }
  if (cnt < 1.0f) cnt = 1.0f;
  g16[b * DD + d] = (h16_t)(s / cnt);
}

// ---- weight transpose + f32->f16:  WT[n*K+k] = W[k*N+n] ----
__global__ void k_wt(const float* __restrict__ W, h16_t* __restrict__ WT, int K, int N) {
  const int t = blockIdx.x * 256 + threadIdx.x;
  if (t >= K * N) return;
  const int n = t / K, k = t - n * K;
  WT[(size_t)n * K + k] = (h16_t)W[(size_t)k * N + n];
}

// ---- plain f32->f16 convert ----
__global__ void k_cvt(const float* __restrict__ X, h16_t* __restrict__ Y, int n) {
  const int t = blockIdx.x * 256 + threadIdx.x;
  if (t < n) Y[t] = (h16_t)X[t];
}

// ---------------------------------------------------------------------------
static inline void gemm(hipStream_t s, const h16_t* A, const h16_t* BT,
                        const float* bias, const float* res,
                        const unsigned char* mask, float* o32, h16_t* o16,
                        int M, int N, int Kd, int act)
{
  dim3 grid(N / 64, M / 128);   // wave = 32x64 tile, 4 waves/block
  k_gemm<<<grid, 128, 0, s>>>(A, BT, bias, res, mask, o32, o16, M, N, Kd, act);
}

extern "C" void kernel_launch(void* const* d_in, const int* in_sizes, int n_in,
                              void* d_out, int out_size, void* d_ws, size_t ws_size,
                              hipStream_t stream)
{
  (void)in_sizes; (void)n_in; (void)out_size; (void)ws_size;
  const float* nodes   = (const float*)d_in[0];
  const unsigned char* mask = (const unsigned char*)d_in[1]; // jnp.bool_ = 1 byte
  const float* emb_w1  = (const float*)d_in[2];
  const float* emb_b1  = (const float*)d_in[3];
  const float* ln1_g   = (const float*)d_in[4];
  const float* ln1_b   = (const float*)d_in[5];
  const float* emb_w2  = (const float*)d_in[6];
  const float* emb_b2  = (const float*)d_in[7];
  const float* ln2_g   = (const float*)d_in[8];
  const float* ln2_b   = (const float*)d_in[9];
  const float* fc1_w   = (const float*)d_in[10];
  const float* fc1_b   = (const float*)d_in[11];
  const float* conv_w  = (const float*)d_in[12];
  const float* conv_b  = (const float*)d_in[13];
  const float* fc2_w   = (const float*)d_in[14];
  const float* fc2_b   = (const float*)d_in[15];
  const float* f1_w    = (const float*)d_in[16];
  const float* f1_b    = (const float*)d_in[17];
  const float* f2_w    = (const float*)d_in[18];
  const float* f2_b    = (const float*)d_in[19];
  const float* out_w1  = (const float*)d_in[20];
  const float* out_b1  = (const float*)d_in[21];
  const float* out_w2  = (const float*)d_in[22];
  const float* out_b2  = (const float*)d_in[23];
  float* out = (float*)d_out;

  // ---- workspace carve-up (256B aligned) ----
  char* ws = (char*)d_ws;
  size_t off = 0;
  auto alloc = [&](size_t bytes) -> char* {
    char* p = ws + off; off += (bytes + 255) & ~(size_t)255; return p;
  };
  h16_t* nodes16 = (h16_t*)alloc((size_t)MROWS * P2D * 2);
  float* x32     = (float*)alloc((size_t)MROWS * DD * 4);
  h16_t* x16     = (h16_t*)alloc((size_t)MROWS * DD * 2);
  float* h32     = (float*)alloc((size_t)MROWS * DD * 4);
  h16_t* h16b    = (h16_t*)alloc((size_t)MROWS * DD * 2);
  float* sq      = (float*)alloc((size_t)MROWS * 4);
  char*  R0      = alloc((size_t)MROWS * 256 * 4);   // t256_32 | d2 | u16 (time-shared)
  h16_t* R1      = (h16_t*)alloc((size_t)MROWS * 256 * 2); // t256_16 | cat16
  int*   idx     = (int*)alloc((size_t)MROWS * KNN * 4);
  h16_t* g16     = (h16_t*)alloc((size_t)BB * DD * 2);
  h16_t* og16    = (h16_t*)alloc((size_t)BB * 512 * 2);
  h16_t* emb_w1T = (h16_t*)alloc((size_t)P2D * 256 * 2);
  h16_t* emb_w2T = (h16_t*)alloc((size_t)256 * DD * 2);
  h16_t* fc1T    = (h16_t*)alloc((size_t)LL * DD * DD * 2);
  h16_t* convT   = (h16_t*)alloc((size_t)LL * 2 * DD * DD * 2);
  h16_t* fc2T    = (h16_t*)alloc((size_t)LL * DD * DD * 2);
  h16_t* f1T     = (h16_t*)alloc((size_t)LL * DD * FFND * 2);
  h16_t* f2T     = (h16_t*)alloc((size_t)LL * FFND * DD * 2);
  h16_t* ow1T    = (h16_t*)alloc((size_t)DD * 512 * 2);
  h16_t* ow2T    = (h16_t*)alloc((size_t)512 * 2048 * 2);

  float* t256_32 = (float*)R0;          // [M,256] embed intermediate
  float* d2      = (float*)R0;          // [B,256,256] distances
  h16_t* u16     = (h16_t*)R0;          // [M,512] FFN hidden (f16)
  h16_t* t256_16 = R1;                  // [M,256] f16
  h16_t* cat16   = R1;                  // [M,256] f16 [h, mrel]

  auto wt = [&](const float* W, h16_t* WT, int K, int N) {
    k_wt<<<(K * N + 255) / 256, 256, 0, stream>>>(W, WT, K, N);
  };

  // ---- weight prep (re-run every call; deterministic) ----
  wt(emb_w1, emb_w1T, P2D, 256);
  wt(emb_w2, emb_w2T, 256, DD);
  for (int l = 0; l < LL; ++l) {
    wt(fc1_w  + (size_t)l * DD * DD,      fc1T  + (size_t)l * DD * DD,      DD,      DD);
    wt(conv_w + (size_t)l * 2 * DD * DD,  convT + (size_t)l * 2 * DD * DD,  2 * DD,  DD);
    wt(fc2_w  + (size_t)l * DD * DD,      fc2T  + (size_t)l * DD * DD,      DD,      DD);
    wt(f1_w   + (size_t)l * DD * FFND,    f1T   + (size_t)l * DD * FFND,    DD,      FFND);
    wt(f2_w   + (size_t)l * FFND * DD,    f2T   + (size_t)l * FFND * DD,    FFND,    DD);
  }
  wt(out_w1, ow1T, DD, 512);
  wt(out_w2, ow2T, 512, 2048);
  k_cvt<<<(MROWS * P2D + 255) / 256, 256, 0, stream>>>(nodes, nodes16, MROWS * P2D);

  // ---- node embed MLP ----
  gemm(stream, nodes16, emb_w1T, emb_b1, nullptr, nullptr, t256_32, nullptr,
       MROWS, 256, P2D, 0);
  k_ln<<<(MROWS + 3) / 4, 128, 0, stream>>>(t256_32, ln1_g, ln1_b, nullptr,
                                            nullptr, t256_16, MROWS, 256, 1);
  gemm(stream, t256_16, emb_w2T, emb_b2, nullptr, nullptr, h32, nullptr,
       MROWS, DD, 256, 0);
  k_ln<<<(MROWS + 3) / 4, 128, 0, stream>>>(h32, ln2_g, ln2_b, mask,
                                            x32, x16, MROWS, DD, 0);

  // ---- ViG blocks ----
  for (int l = 0; l < LL; ++l) {
    // h = x @ fc1 + b
    gemm(stream, x16, fc1T + (size_t)l * DD * DD, fc1_b + l * DD,
         nullptr, nullptr, h32, h16b, MROWS, DD, DD, 0);
    // masked kNN in feature space
    k_sqnorm<<<MROWS / 256, 256, 0, stream>>>(h32, sq, MROWS);
    {
      dim3 grid(NN / 16, (NN / 16) / 4, BB);
      k_gram_d2<<<grid, 128, 0, stream>>>(h16b, sq, mask, d2);
    }
    k_topk<<<MROWS / 256, 256, 0, stream>>>(d2, idx, MROWS);
    // max-relative aggregation + concat to f16
    k_mrel<<<MROWS, 128, 0, stream>>>(h32, idx, mask, cat16);
    // h = lrelu(cat @ conv_w + b)
    gemm(stream, cat16, convT + (size_t)l * 2 * DD * DD, conv_b + l * DD,
         nullptr, nullptr, nullptr, h16b, MROWS, DD, 2 * DD, 1);
    // x = h @ fc2 + b + sc   (in-place residual: read res before write)
    gemm(stream, h16b, fc2T + (size_t)l * DD * DD, fc2_b + l * DD,
         x32, nullptr, x32, x16, MROWS, DD, DD, 0);
    // FFN: u = lrelu(x @ f1 + b); x = (u @ f2 + b + sc) * mf
    gemm(stream, x16, f1T + (size_t)l * DD * FFND, f1_b + l * FFND,
         nullptr, nullptr, nullptr, u16, MROWS, FFND, DD, 1);
    gemm(stream, u16, f2T + (size_t)l * FFND * DD, f2_b + l * DD,
         x32, mask, x32, x16, MROWS, DD, FFND, 0);
  }

  // ---- masked mean pool + head MLP ----
  k_pool<<<BB, 128, 0, stream>>>(x32, mask, g16);
  gemm(stream, g16, ow1T, out_b1, nullptr, nullptr, nullptr, og16, BB, 512, DD, 1);
  gemm(stream, og16, ow2T, out_b2, nullptr, nullptr, out, nullptr, BB, 2048, 512, 0);
}
